// AttentiveDensenet_31233002176654
// MI455X (gfx1250) — matrix operand order repaired and checked by reading, not compiled, not run on hardware
//
#include <hip/hip_runtime.h>
#include <hip/hip_bf16.h>

typedef __attribute__((ext_vector_type(16))) _Float16 v16h;
typedef __attribute__((ext_vector_type(8)))  float    v8f;
typedef __attribute__((ext_vector_type(4)))  int      v4i;

union HF { uint4 u[2]; v16h h; };

// Problem constants (from reference): B=8, C=256, H=W=32, NH=8, K=V=64, L=4
constexpr int NPIX = 8192;          // B*H*W
constexpr int CDIM = 256;
constexpr int KH   = 512;           // NH*K = NH*V
constexpr int LAY  = 4;

// ---------------------------------------------------------------------------
// CDNA5 async global->LDS copy (ASYNCcnt-tracked). Falls back to a synchronous
// VGPR round-trip when the builtin is unavailable on this toolchain.
// ---------------------------------------------------------------------------
#if __has_builtin(__builtin_amdgcn_global_load_async_to_lds_b128)
#define HAVE_ASYNC_LDS 1
#else
#define HAVE_ASYNC_LDS 0
#endif

__device__ __forceinline__ void g2l_copy16(const _Float16* g, _Float16* l) {
#if HAVE_ASYNC_LDS
  __builtin_amdgcn_global_load_async_to_lds_b128(
      (__attribute__((address_space(1))) v4i*)(uintptr_t)g,
      (__attribute__((address_space(3))) v4i*)(unsigned int)(uintptr_t)l,
      0, 0);
#else
  *(uint4*)l = *(const uint4*)g;
#endif
}

__device__ __forceinline__ void wait_async_lds() {
#if HAVE_ASYNC_LDS
#if __has_builtin(__builtin_amdgcn_s_wait_asynccnt)
  __builtin_amdgcn_s_wait_asynccnt(0);
#else
  asm volatile("s_wait_asynccnt 0x0" ::: "memory");
#endif
#endif
}

// ---------------------------------------------------------------------------
// conv1x1 as GEMM: out[O,NPIX] = A[O,Cin](f16) * Bm[Cin,NPIX](f16) + bias
// 256 thr = 8 waves; block tile 128x128, wave tile 64x32, K-step 32.
// Double-buffered LDS: async A tile copies + register-prefetched B transpose.
// ---------------------------------------------------------------------------
__global__ __launch_bounds__(256) void gemm1x1_wmma(
    const _Float16* __restrict__ A, const float* __restrict__ bias,
    const _Float16* __restrict__ Bm, float* __restrict__ out, int Cin)
{
  __shared__ _Float16 As[2][128][40];
  __shared__ _Float16 Bs[2][128][40];
  const int tid  = threadIdx.x;
  const int lane = tid & 31;
  const int wave = tid >> 5;
  const int wM = wave >> 2;   // 0..1
  const int wN = wave & 3;    // 0..3
  const int m0 = blockIdx.y * 128;
  const int n0 = blockIdx.x * 128;

  v8f acc[4][2];
  #pragma unroll
  for (int i = 0; i < 4; ++i)
    #pragma unroll
    for (int j = 0; j < 2; ++j) { v8f z = {}; acc[i][j] = z; }

  const int ar = tid >> 1;            // A row 0..127
  const int as = (tid & 1) * 16;      // A half-offset 0/16
  const int bk = tid >> 3;            // B k-row 0..31
  const int bs = tid & 7;             // B col segment 0..7

  const int koffA = (lane < 16) ? 0 : 8;
  const int rA = lane & 15;
  const int kB = (lane >> 4) * 8;
  const int cB = lane & 15;

  // ---- prologue: stage K-step 0 into buffer 0 ----
  g2l_copy16(A + (size_t)(m0 + ar) * Cin + as,     &As[0][ar][as]);
  g2l_copy16(A + (size_t)(m0 + ar) * Cin + as + 8, &As[0][ar][as + 8]);
  {
    __align__(16) _Float16 t0[16];
    const uint4* src = (const uint4*)(Bm + (size_t)bk * NPIX + n0 + bs * 16);
    *(uint4*)(t0)     = src[0];
    *(uint4*)(t0 + 8) = src[1];
    #pragma unroll
    for (int j = 0; j < 16; ++j) Bs[0][bs * 16 + j][bk] = t0[j];
  }
  wait_async_lds();
  __syncthreads();

  const int S = Cin / 32;
  int cur = 0;
  for (int ks = 0; ks < S; ++ks) {
    const int nxt = cur ^ 1;
    const bool pf = (ks + 1 < S);
    __align__(16) _Float16 t0[16];
    if (pf) {   // prefetch K-step ks+1: A async into LDS, B into registers
      const int k1 = (ks + 1) * 32;
      g2l_copy16(A + (size_t)(m0 + ar) * Cin + k1 + as,     &As[nxt][ar][as]);
      g2l_copy16(A + (size_t)(m0 + ar) * Cin + k1 + as + 8, &As[nxt][ar][as + 8]);
      const uint4* src = (const uint4*)(Bm + (size_t)(k1 + bk) * NPIX + n0 + bs * 16);
      *(uint4*)(t0)     = src[0];
      *(uint4*)(t0 + 8) = src[1];
    }

    HF bfrag[2];
    #pragma unroll
    for (int nj = 0; nj < 2; ++nj) {
      const int col = wN * 32 + nj * 16 + cB;
      bfrag[nj].u[0] = *(const uint4*)(&Bs[cur][col][kB]);
      bfrag[nj].u[1] = *(const uint4*)(&Bs[cur][col][kB + 16]);
    }
    #pragma unroll
    for (int mi = 0; mi < 4; ++mi) {
      const int row = wM * 64 + mi * 16 + rA;
      HF afrag;
      afrag.u[0] = *(const uint4*)(&As[cur][row][koffA]);
      afrag.u[1] = *(const uint4*)(&As[cur][row][16 + koffA]);
      #pragma unroll
      for (int nj = 0; nj < 2; ++nj)
        acc[mi][nj] = __builtin_amdgcn_wmma_f32_16x16x32_f16(
            false, afrag.h, false, bfrag[nj].h, (short)0, acc[mi][nj], false, false);
    }

    if (pf) {   // transpose-scatter the prefetched B tile into the other buffer
      #pragma unroll
      for (int j = 0; j < 16; ++j) Bs[nxt][bs * 16 + j][bk] = t0[j];
      wait_async_lds();
    }
    __syncthreads();
    cur = nxt;
  }

  #pragma unroll
  for (int mi = 0; mi < 4; ++mi)
    #pragma unroll
    for (int nj = 0; nj < 2; ++nj)
      #pragma unroll
      for (int r = 0; r < 8; ++r) {
        const int row = m0 + wM * 64 + mi * 16 + ((lane < 16) ? r : 8 + r);
        const int col = n0 + wN * 32 + nj * 16 + (lane & 15);
        out[(size_t)row * NPIX + col] = acc[mi][nj][r] + bias[row];
      }
}

// ---------------------------------------------------------------------------
// 3x3 conv (SAME, pad 1) as implicit GEMM over K' = 9*Cin.
// A = weights packed [O, 9, Cin] f16 (tap-major). B loader shifts/masks.
// Same double-buffered pipeline as gemm1x1.
// ---------------------------------------------------------------------------
__device__ __forceinline__ void conv_gatherB(
    const _Float16* __restrict__ ip, int n0, int bs, int dy, int dx, _Float16* t0)
{
  #pragma unroll
  for (int j = 0; j < 16; ++j) {
    const int n  = n0 + bs * 16 + j;
    const int hw = n & 1023;
    const int b  = n >> 10;
    const int h  = hw >> 5, w = hw & 31;
    const int hs = h + dy, ws = w + dx;
    _Float16 v = (_Float16)0.f;
    if ((unsigned)hs < 32u && (unsigned)ws < 32u)
      v = ip[(b << 10) + (hs << 5) + ws];
    t0[j] = v;
  }
}

__global__ __launch_bounds__(256) void conv3x3_wmma(
    const _Float16* __restrict__ A, const float* __restrict__ bias,
    const _Float16* __restrict__ inh, float* __restrict__ out, int Cin)
{
  __shared__ _Float16 As[2][128][40];
  __shared__ _Float16 Bs[2][128][40];
  const int tid  = threadIdx.x;
  const int lane = tid & 31;
  const int wave = tid >> 5;
  const int wM = wave >> 2;
  const int wN = wave & 3;
  const int m0 = blockIdx.y * 128;
  const int n0 = blockIdx.x * 128;
  const int Ktot = 9 * Cin;

  v8f acc[4][2];
  #pragma unroll
  for (int i = 0; i < 4; ++i)
    #pragma unroll
    for (int j = 0; j < 2; ++j) { v8f z = {}; acc[i][j] = z; }

  const int ar = tid >> 1;
  const int as = (tid & 1) * 16;
  const int bk = tid >> 3;
  const int bs = tid & 7;

  const int koffA = (lane < 16) ? 0 : 8;
  const int rA = lane & 15;
  const int kB = (lane >> 4) * 8;
  const int cB = lane & 15;

  // ---- prologue: K-step 0 (tap 0 => dy=-1, dx=-1) into buffer 0 ----
  g2l_copy16(A + (size_t)(m0 + ar) * Ktot + as,     &As[0][ar][as]);
  g2l_copy16(A + (size_t)(m0 + ar) * Ktot + as + 8, &As[0][ar][as + 8]);
  {
    __align__(16) _Float16 t0[16];
    conv_gatherB(inh + (size_t)bk * NPIX, n0, bs, -1, -1, t0);
    #pragma unroll
    for (int j = 0; j < 16; ++j) Bs[0][bs * 16 + j][bk] = t0[j];
  }
  wait_async_lds();
  __syncthreads();

  const int S = Ktot / 32;
  int cur = 0;
  for (int ks = 0; ks < S; ++ks) {
    const int nxt = cur ^ 1;
    const bool pf = (ks + 1 < S);
    __align__(16) _Float16 t0[16];
    if (pf) {
      const int k1   = (ks + 1) * 32;
      const int tap1 = k1 / Cin;           // Cin % 32 == 0 -> tap fixed per step
      const int c0   = k1 - tap1 * Cin;
      g2l_copy16(A + (size_t)(m0 + ar) * Ktot + k1 + as,     &As[nxt][ar][as]);
      g2l_copy16(A + (size_t)(m0 + ar) * Ktot + k1 + as + 8, &As[nxt][ar][as + 8]);
      conv_gatherB(inh + (size_t)(c0 + bk) * NPIX, n0, bs,
                   tap1 / 3 - 1, tap1 % 3 - 1, t0);
    }

    HF bfrag[2];
    #pragma unroll
    for (int nj = 0; nj < 2; ++nj) {
      const int col = wN * 32 + nj * 16 + cB;
      bfrag[nj].u[0] = *(const uint4*)(&Bs[cur][col][kB]);
      bfrag[nj].u[1] = *(const uint4*)(&Bs[cur][col][kB + 16]);
    }
    #pragma unroll
    for (int mi = 0; mi < 4; ++mi) {
      const int row = wM * 64 + mi * 16 + rA;
      HF afrag;
      afrag.u[0] = *(const uint4*)(&As[cur][row][koffA]);
      afrag.u[1] = *(const uint4*)(&As[cur][row][16 + koffA]);
      #pragma unroll
      for (int nj = 0; nj < 2; ++nj)
        acc[mi][nj] = __builtin_amdgcn_wmma_f32_16x16x32_f16(
            false, afrag.h, false, bfrag[nj].h, (short)0, acc[mi][nj], false, false);
    }

    if (pf) {
      #pragma unroll
      for (int j = 0; j < 16; ++j) Bs[nxt][bs * 16 + j][bk] = t0[j];
      wait_async_lds();
    }
    __syncthreads();
    cur = nxt;
  }

  #pragma unroll
  for (int mi = 0; mi < 4; ++mi)
    #pragma unroll
    for (int nj = 0; nj < 2; ++nj)
      #pragma unroll
      for (int r = 0; r < 8; ++r) {
        const int row = m0 + wM * 64 + mi * 16 + ((lane < 16) ? r : 8 + r);
        const int col = n0 + wN * 32 + nj * 16 + (lane & 15);
        out[(size_t)row * NPIX + col] = acc[mi][nj][r] + bias[row];
      }
}

// ---------------------------------------------------------------------------
// Sparse top-k attention over the growing key/value history (T = l+2 incl. zero)
// ---------------------------------------------------------------------------
__global__ __launch_bounds__(256) void attention_kernel(
    const float* __restrict__ q, const float* __restrict__ khist,
    const float* __restrict__ vhist, _Float16* __restrict__ oh, int l)
{
  const int tid  = blockIdx.x * 256 + threadIdx.x;   // 0..65535
  const int n    = tid & (NPIX - 1);
  const int head = tid >> 13;
  const size_t base = (size_t)head * 64 * NPIX + n;
  const size_t LSTRIDE = (size_t)KH * NPIX;

  float qv[64];
  #pragma unroll
  for (int d = 0; d < 64; ++d) qv[d] = q[base + (size_t)d * NPIX];

  const int T = l + 2;
  float sc[5];
  for (int t = 0; t <= l; ++t) {
    const float* kp = khist + (size_t)t * LSTRIDE + base;
    float s = 0.f;
    #pragma unroll
    for (int d = 0; d < 64; ++d) s += qv[d] * kp[(size_t)d * NPIX];
    sc[t] = s * 0.125f;                 // /sqrt(K)
  }
  sc[T - 1] = 0.f;                      // appended zero key

  float mx = sc[0];
  for (int t = 1; t < T; ++t) mx = fmaxf(mx, sc[t]);
  float at[5], ssum = 0.f;
  for (int t = 0; t < T; ++t) { at[t] = __expf(sc[t] - mx); ssum += at[t]; }
  const float inv_s = 1.f / ssum;
  for (int t = 0; t < T; ++t) at[t] *= inv_s;

  if (T > 4) {                          // SparseAttention: threshold at 4th-largest
    float m1 = 1e30f, m2 = 1e30f;
    for (int t = 0; t < T; ++t) {
      const float v = at[t];
      if (v < m1) { m2 = m1; m1 = v; } else if (v < m2) { m2 = v; }
    }
    const float delta = m2 + 1e-7f;
    float wsum = 0.f;
    for (int t = 0; t < T; ++t) { at[t] = fmaxf(at[t] - delta, 0.f); wsum += at[t]; }
    const float inv = 1.f / (wsum + 1e-7f);
    for (int t = 0; t < T; ++t) at[t] *= inv;
  }

  for (int d = 0; d < 64; ++d) {
    float a = 0.f;
    for (int t = 0; t <= l; ++t)
      a += at[t] * vhist[(size_t)t * LSTRIDE + base + (size_t)d * NPIX];
    oh[base + (size_t)d * NPIX] = (_Float16)a;
  }
}

// ---------------------------------------------------------------------------
// BatchNorm (training mode): per-channel mean / biased var -> scale, shift
// ---------------------------------------------------------------------------
__global__ __launch_bounds__(256) void bn_stats_kernel(
    const float* __restrict__ y, const float* __restrict__ g,
    const float* __restrict__ b, float* __restrict__ scale,
    float* __restrict__ shift)
{
  const int c = blockIdx.x;
  const float* yp = y + (size_t)c * NPIX;
  float s = 0.f, sq = 0.f;
  for (int i = threadIdx.x; i < NPIX; i += 256) {
    const float v = yp[i];
    s += v; sq += v * v;
  }
  __shared__ float sh1[256], sh2[256];
  sh1[threadIdx.x] = s; sh2[threadIdx.x] = sq;
  __syncthreads();
  for (int off = 128; off > 0; off >>= 1) {
    if (threadIdx.x < off) {
      sh1[threadIdx.x] += sh1[threadIdx.x + off];
      sh2[threadIdx.x] += sh2[threadIdx.x + off];
    }
    __syncthreads();
  }
  if (threadIdx.x == 0) {
    const float mean = sh1[0] * (1.f / NPIX);
    const float var  = sh2[0] * (1.f / NPIX) - mean * mean;
    const float sc   = g[c] * rsqrtf(var + 1e-5f);
    scale[c] = sc;
    shift[c] = b[c] - mean * sc;
  }
}

__global__ __launch_bounds__(256) void bn_relu_kernel(
    const float* __restrict__ y, const float* __restrict__ scale,
    const float* __restrict__ shift, _Float16* __restrict__ h1h)
{
  const int i = blockIdx.x * 256 + threadIdx.x;   // over CDIM*NPIX
  const int c = i >> 13;
  const float v = y[i] * scale[c] + shift[c];
  h1h[i] = (_Float16)fmaxf(v, 0.f);
}

__global__ __launch_bounds__(256) void residual_kernel(
    float* __restrict__ x_cn, _Float16* __restrict__ xh,
    const float* __restrict__ h2, const float* __restrict__ gammas, int l)
{
  const int i = blockIdx.x * 256 + threadIdx.x;
  const float g = gammas[l];
  const float v = x_cn[i] + h2[i] * g;
  x_cn[i] = v;
  xh[i]   = (_Float16)v;
}

// ---- layout / precision packing kernels -----------------------------------
__global__ __launch_bounds__(256) void pack_x_kernel(
    const float* __restrict__ x, float* __restrict__ x_cn, _Float16* __restrict__ xh)
{
  const int i = blockIdx.x * 256 + threadIdx.x;   // i = c*8192 + b*1024 + hw
  const int c = i >> 13;
  const int rem = i & 8191;
  const int b = rem >> 10;
  const int hw = rem & 1023;
  const float v = x[((size_t)b * CDIM + c) * 1024 + hw];
  x_cn[i] = v;
  xh[i]   = (_Float16)v;
}

__global__ __launch_bounds__(256) void unpack_x_kernel(
    const float* __restrict__ x_cn, float* __restrict__ out)
{
  const int i = blockIdx.x * 256 + threadIdx.x;   // NCHW index
  const int b = i >> 18;
  const int c = (i >> 10) & 255;
  const int hw = i & 1023;
  out[i] = x_cn[((size_t)c << 13) + (b << 10) + hw];
}

__global__ __launch_bounds__(256) void f32_to_f16_kernel(
    const float* __restrict__ src, _Float16* __restrict__ dst, int n)
{
  const int i = blockIdx.x * 256 + threadIdx.x;
  if (i < n) dst[i] = (_Float16)src[i];
}

__global__ __launch_bounds__(256) void pack_w3_kernel(
    const float* __restrict__ w, _Float16* __restrict__ wh, int Cin)
{
  const int i = blockIdx.x * 256 + threadIdx.x;   // out: o*(9*Cin)+tap*Cin+c
  const int tot = CDIM * Cin * 9;
  if (i >= tot) return;
  const int o = i / (9 * Cin);
  const int r = i - o * 9 * Cin;
  const int tap = r / Cin;
  const int c = r - tap * Cin;
  wh[i] = (_Float16)w[((size_t)o * Cin + c) * 9 + tap];
}

// ---------------------------------------------------------------------------
extern "C" void kernel_launch(void* const* d_in, const int* in_sizes, int n_in,
                              void* d_out, int out_size, void* d_ws, size_t ws_size,
                              hipStream_t stream)
{
  (void)in_sizes; (void)n_in; (void)out_size; (void)ws_size;
  const float* x      = (const float*)d_in[0];
  const float* kw     = (const float*)d_in[1];
  const float* kb     = (const float*)d_in[2];
  const float* qw     = (const float*)d_in[3];
  const float* qb     = (const float*)d_in[4];
  const float* vw     = (const float*)d_in[5];
  const float* vb     = (const float*)d_in[6];
  const float* ow1    = (const float*)d_in[7];
  const float* ob1    = (const float*)d_in[8];
  const float* bn_g   = (const float*)d_in[9];
  const float* bn_b   = (const float*)d_in[10];
  const float* ow2    = (const float*)d_in[11];
  const float* ob2    = (const float*)d_in[12];
  const float* gammas = (const float*)d_in[13];
  float* out = (float*)d_out;

  // workspace carve-up
  char* ws = (char*)d_ws;
  size_t off = 0;
  auto alloc = [&](size_t bytes) {
    void* p = ws + off;
    off += (bytes + 255) & ~(size_t)255;
    return p;
  };
  const size_t CN  = (size_t)CDIM * NPIX;   // 2M
  const size_t KHN = (size_t)KH * NPIX;     // 4M

  float*    x_cn  = (float*)   alloc(CN * 4);
  _Float16* xh    = (_Float16*)alloc(CN * 2);
  float*    qbuf  = (float*)   alloc(KHN * 4);
  float*    khist = (float*)   alloc((size_t)LAY * KHN * 4);
  float*    vhist = (float*)   alloc((size_t)LAY * KHN * 4);
  _Float16* oh    = (_Float16*)alloc(KHN * 2);
  float*    ybuf  = (float*)   alloc(CN * 4);
  _Float16* h1h   = (_Float16*)alloc(CN * 2);
  float*    h2    = (float*)   alloc(CN * 4);
  _Float16* wk_h  = (_Float16*)alloc((size_t)KH * CDIM * 2);
  _Float16* wq_h  = (_Float16*)alloc((size_t)KH * CDIM * 2);
  _Float16* wv_h  = (_Float16*)alloc((size_t)KH * CDIM * 2);
  _Float16* w3a_h = (_Float16*)alloc((size_t)CDIM * 9 * KH * 2);
  _Float16* w3b_h = (_Float16*)alloc((size_t)CDIM * 9 * CDIM * 2);
  float*    bnsc  = (float*)   alloc(CDIM * 4);
  float*    bnsh  = (float*)   alloc(CDIM * 4);

  const dim3 blk(256);
  const dim3 gElemCN((unsigned)(CN / 256));
  const int  nWkqv = KH * CDIM;

  pack_x_kernel<<<gElemCN, blk, 0, stream>>>(x, x_cn, xh);

  for (int l = 0; l < LAY; ++l) {
    // k, v, q projections (conv1x1 as WMMA GEMM)
    f32_to_f16_kernel<<<(nWkqv + 255) / 256, blk, 0, stream>>>(kw + (size_t)l * nWkqv, wk_h, nWkqv);
    f32_to_f16_kernel<<<(nWkqv + 255) / 256, blk, 0, stream>>>(vw + (size_t)l * nWkqv, wv_h, nWkqv);
    f32_to_f16_kernel<<<(nWkqv + 255) / 256, blk, 0, stream>>>(qw + (size_t)l * nWkqv, wq_h, nWkqv);
    gemm1x1_wmma<<<dim3(NPIX / 128, KH / 128), blk, 0, stream>>>(wk_h, kb + (size_t)l * KH, xh, khist + (size_t)l * KHN, CDIM);
    gemm1x1_wmma<<<dim3(NPIX / 128, KH / 128), blk, 0, stream>>>(wv_h, vb + (size_t)l * KH, xh, vhist + (size_t)l * KHN, CDIM);
    gemm1x1_wmma<<<dim3(NPIX / 128, KH / 128), blk, 0, stream>>>(wq_h, qb + (size_t)l * KH, xh, qbuf, CDIM);

    // sparse top-k attention over the history
    attention_kernel<<<(NPIX * 8) / 256, blk, 0, stream>>>(qbuf, khist, vhist, oh, l);

    // conv3x3 (VH=512 -> C=256) + BN + ReLU
    {
      const int tot = CDIM * KH * 9;
      pack_w3_kernel<<<(tot + 255) / 256, blk, 0, stream>>>(ow1 + (size_t)l * tot, w3a_h, KH);
    }
    conv3x3_wmma<<<dim3(NPIX / 128, CDIM / 128), blk, 0, stream>>>(w3a_h, ob1 + (size_t)l * CDIM, oh, ybuf, KH);
    bn_stats_kernel<<<CDIM, blk, 0, stream>>>(ybuf, bn_g + (size_t)l * CDIM, bn_b + (size_t)l * CDIM, bnsc, bnsh);
    bn_relu_kernel<<<gElemCN, blk, 0, stream>>>(ybuf, bnsc, bnsh, h1h);

    // conv3x3 (C=256 -> C=256), residual with gamma
    {
      const int tot = CDIM * CDIM * 9;
      pack_w3_kernel<<<(tot + 255) / 256, blk, 0, stream>>>(ow2 + (size_t)l * tot, w3b_h, CDIM);
    }
    conv3x3_wmma<<<dim3(NPIX / 128, CDIM / 128), blk, 0, stream>>>(w3b_h, ob2 + (size_t)l * CDIM, h1h, h2, CDIM);
    residual_kernel<<<gElemCN, blk, 0, stream>>>(x_cn, xh, h2, gammas, l);
  }

  unpack_x_kernel<<<gElemCN, blk, 0, stream>>>(x_cn, out);
}
